// MoELayer_46282567582071
// MI455X (gfx1250) — matrix operand order repaired and checked by reading, not compiled, not run on hardware
//
#include <hip/hip_runtime.h>
#include <hip/hip_bf16.h>
#include <stdint.h>

// ---------------------------------------------------------------------------
// MoE layer for MI455X (gfx1250).  B=4,S=4096,D=1024,E=8,K=2,H=2048.
// Reference scatters by EXPERT id => only output rows 0..7 are non-zero, and
// the linear down-projection commutes with the sum: accumulate per-expert
// hsum[e] = sum_j silu(y_j@Wg[e]) * (y_j@Wu[e]), then 8 GEMVs with w_down.
// Dominant GEMM: v_wmma_f32_16x16x32_bf16, full-depth bf16 weight tiles
// resident in CDNA5's 320KB WGP LDS (288KB dynamic LDS per workgroup).
// LDS is stored pre-swizzled in WMMA fragment order so every fragment load
// is two ds_load_b128 instead of 16 ds_load_u16.
// ---------------------------------------------------------------------------

#define B_DIM   4
#define S_DIM   4096
#define D_DIM   1024
#define E_DIM   8
#define H_DIM   2048
#define T_TOK   (B_DIM * S_DIM)      // 16384 tokens
#define TILE_N  64                   // H columns per workgroup (4 waves x 16)
#define OUT_N   ((size_t)T_TOK * D_DIM + 1)

typedef __attribute__((ext_vector_type(16))) __bf16 v16bf;
typedef __attribute__((ext_vector_type(8)))  float  v8f;

__device__ __forceinline__ __bf16 f2bf(float f) {
  unsigned u = __builtin_bit_cast(unsigned, f);
  unsigned r = (u + 0x7FFFu + ((u >> 16) & 1u)) >> 16;   // round-to-nearest-even
  unsigned short h = (unsigned short)r;
  return __builtin_bit_cast(__bf16, h);
}

// ---------------------------------------------------------------------------
// Kernel 0: zero output + routing counters (harness poisons buffers with 0xAA)
// ---------------------------------------------------------------------------
__global__ void moe_zero(float* __restrict__ out, size_t n_out,
                         int* __restrict__ cnt, int* __restrict__ f_cnt,
                         float* __restrict__ p_sum) {
  size_t i = (size_t)blockIdx.x * blockDim.x + threadIdx.x;
  size_t stride = (size_t)gridDim.x * blockDim.x;
  for (size_t j = i; j < n_out; j += stride) out[j] = 0.0f;
  if (i < E_DIM) { cnt[i] = 0; f_cnt[i] = 0; p_sum[i] = 0.0f; }
}

// ---------------------------------------------------------------------------
// Kernel 1: gating.  One wave32 per token: logits = x_row @ gate_w [1024x8],
// full softmax -> p_sum (aux loss), top-2 -> softmax weights, append token to
// per-expert gather lists.
// ---------------------------------------------------------------------------
__global__ void moe_gate(const float* __restrict__ x,
                         const float* __restrict__ gate_w,
                         int* __restrict__ cnt, int* __restrict__ f_cnt,
                         float* __restrict__ p_sum,
                         int* __restrict__ list, float* __restrict__ wl) {
  const int wave = threadIdx.x >> 5;       // 8 waves per block
  const int lane = threadIdx.x & 31;
  const int t = blockIdx.x * 8 + wave;

  float acc[E_DIM];
#pragma unroll
  for (int e = 0; e < E_DIM; ++e) acc[e] = 0.0f;

  const float* xr = x + (size_t)t * D_DIM;
#pragma unroll 4
  for (int i = 0; i < D_DIM / 32; ++i) {
    int d = i * 32 + lane;
    float xv = xr[d];
    const float* gw = gate_w + (size_t)d * E_DIM;
#pragma unroll
    for (int e = 0; e < E_DIM; ++e) acc[e] += xv * gw[e];
  }
#pragma unroll
  for (int e = 0; e < E_DIM; ++e) {
#pragma unroll
    for (int off = 16; off > 0; off >>= 1)
      acc[e] += __shfl_down(acc[e], off, 32);
  }

  if (lane == 0) {
    // full softmax for p_i (aux loss)
    float mx = acc[0];
#pragma unroll
    for (int e = 1; e < E_DIM; ++e) mx = fmaxf(mx, acc[e]);
    float p[E_DIM], s = 0.0f;
#pragma unroll
    for (int e = 0; e < E_DIM; ++e) { p[e] = __expf(acc[e] - mx); s += p[e]; }
    float inv = __builtin_amdgcn_rcpf(s);
#pragma unroll
    for (int e = 0; e < E_DIM; ++e) atomicAdd(&p_sum[e], p[e] * inv);

    // top-2 (ties keep lowest index, matching lax.top_k)
    int i0 = 0;
#pragma unroll
    for (int e = 1; e < E_DIM; ++e) if (acc[e] > acc[i0]) i0 = e;
    int i1 = (i0 == 0) ? 1 : 0;
#pragma unroll
    for (int e = 0; e < E_DIM; ++e)
      if (e != i0 && acc[e] > acc[i1]) i1 = e;

    float eb = __expf(acc[i1] - acc[i0]);   // acc[i0] >= acc[i1]
    float w0 = __builtin_amdgcn_rcpf(1.0f + eb);
    float w1 = eb * w0;

    atomicAdd(&f_cnt[i0], 1);
    atomicAdd(&f_cnt[i1], 1);
    int p0 = atomicAdd(&cnt[i0], 1);
    list[i0 * T_TOK + p0] = t;  wl[i0 * T_TOK + p0] = w0;
    int p1 = atomicAdd(&cnt[i1], 1);
    list[i1 * T_TOK + p1] = t;  wl[i1 * T_TOK + p1] = w1;
  }
}

// ---------------------------------------------------------------------------
// Kernel 2: aux loss = sum_e (f_cnt[e]/T)*(p_sum[e]/T) * E^2
// ---------------------------------------------------------------------------
__global__ void moe_aux(const int* __restrict__ f_cnt,
                        const float* __restrict__ p_sum,
                        float* __restrict__ out_aux) {
  int lane = threadIdx.x;
  float v = 0.0f;
  if (lane < E_DIM) {
    float fi = (float)f_cnt[lane] / (float)T_TOK;
    float pi = p_sum[lane] / (float)T_TOK;
    v = fi * pi * (float)(E_DIM * E_DIM);
  }
#pragma unroll
  for (int off = 16; off > 0; off >>= 1) v += __shfl_down(v, off, 32);
  if (lane == 0) *out_aux = v;
}

// ---------------------------------------------------------------------------
// Kernel 3: per-expert fused gate/up GEMM + SiLU*up + row reduction.
//   grid = (H/64, E), block = 128 (4 waves); wave w owns columns w*16..w*16+15
//
// LDS (dynamic, 288KB) stored in WMMA *fragment order*:
//   A  : [D/32 kchunks][32 lanes][16 bf16]   (lane = half*16 + m,
//         lane holds K = half*16 + 0..15, contiguous; 32B contiguous per lane)
//   Bg/Bu: [D/32 kchunks][64 cols][32 bf16]  (lane fragment = 16 contiguous
//         K at kl = half*16; 32B contiguous)
// Every fragment load is a single 32-byte LDS read (2x ds_load_b128).
// ---------------------------------------------------------------------------
__global__ void __launch_bounds__(128)
moe_expert_gemm(const float* __restrict__ x,
                const float* __restrict__ w_gate,
                const float* __restrict__ w_up,
                const int* __restrict__ cnt,
                const int* __restrict__ list,
                const float* __restrict__ wl,
                float* __restrict__ hsum) {
  const int e    = blockIdx.y;
  const int h0   = blockIdx.x * TILE_N;
  const int tid  = threadIdx.x;
  const int lane = tid & 31;
  const int wave = tid >> 5;
  const int l15  = lane & 15;    // A row m / B-C column n within subtile
  const int half = lane >> 4;

  extern __shared__ __align__(32) __bf16 smem[];
  __bf16* Bg = smem;                        // [D/32][64][32]
  __bf16* Bu = Bg + (size_t)D_DIM * TILE_N; // [D/32][64][32]
  __bf16* As = Bu + (size_t)D_DIM * TILE_N; // [D/32][32][16]
  __shared__ int   s_tok[16];
  __shared__ float s_w[16];

  // ---- stage full-depth weight tiles once (bf16, fragment-swizzled) -------
  const float* g = w_gate + (size_t)e * D_DIM * H_DIM + h0;
  const float* u = w_up   + (size_t)e * D_DIM * H_DIM + h0;
  for (int i = tid; i < D_DIM * TILE_N; i += blockDim.x) {
    int d  = i >> 6;            // /TILE_N
    int j  = i & (TILE_N - 1);
    int kc = d >> 5;
    int kl = d & 31;
    int o  = ((kc * TILE_N + j) << 5) + kl;   // [kc][col][kl]
    Bg[o] = f2bf(g[(size_t)d * H_DIM + j]);
    Bu[o] = f2bf(u[(size_t)d * H_DIM + j]);
  }
  __syncthreads();

  const int n = cnt[e];
  const int nsteps = (n + 15) >> 4;
  float colsum = 0.0f;
  const int col = wave * 16 + l15;

  for (int step = 0; step < nsteps; ++step) {
    const int m0 = step << 4;
    if (tid < 16) {
      int m = m0 + tid;
      s_tok[tid] = (m < n) ? list[e * T_TOK + m] : -1;
      s_w[tid]   = (m < n) ? wl[e * T_TOK + m] : 0.0f;
      // prefetch next chunk's routing entries (global_prefetch_b8)
      if (m + 16 < n) {
        __builtin_prefetch(&list[e * T_TOK + m + 16], 0, 0);
        __builtin_prefetch(&wl[e * T_TOK + m + 16], 0, 0);
      }
    }
    __syncthreads();

    // stage A: 16 token rows x D, pre-scaled, bf16, fragment-swizzled
    for (int i = tid; i < 16 * D_DIM; i += blockDim.x) {
      int m  = i >> 10;          // /D_DIM
      int d  = i & (D_DIM - 1);
      int tk = s_tok[m];
      float v = (tk >= 0) ? x[(size_t)tk * D_DIM + d] * s_w[m] : 0.0f;
      int kc = d >> 5;
      int kl = d & 31;
      int ah = kl >> 4;                                // lane half owning kl
      int vv = (kl & 15) >> 1;                         // element pair 0..7
      int pos = 2 * vv + (kl & 1);
      As[((kc * 32 + ah * 16 + m) << 4) + pos] = f2bf(v);
    }
    __syncthreads();

    v8f cg = {};
    v8f cu = {};
#pragma unroll 2
    for (int kc = 0; kc < D_DIM / 32; ++kc) {
      v16bf a  = *(const v16bf*)(As + ((kc * 32 + lane) << 4));
      v16bf bg = *(const v16bf*)(Bg + ((kc * TILE_N + col) << 5) + half * 16);
      v16bf bu = *(const v16bf*)(Bu + ((kc * TILE_N + col) << 5) + half * 16);
      cg = __builtin_amdgcn_wmma_f32_16x16x32_bf16(false, a, false, bg,
                                                   (short)0, cg, false, false);
      cu = __builtin_amdgcn_wmma_f32_16x16x32_bf16(false, a, false, bu,
                                                   (short)0, cu, false, false);
    }

    // h = silu(g) * u; reduce over the 16 token rows of this chunk.
    // C layout: lane holds col n=l15, rows r + half*8 (r=0..7).
    float s = 0.0f;
#pragma unroll
    for (int r = 0; r < 8; ++r) {
      float gv  = cg[r];
      float uv  = cu[r];
      float sig = __builtin_amdgcn_rcpf(1.0f + __expf(-gv));   // fast sigmoid
      s += gv * sig * uv;
    }
    s += __shfl_xor(s, 16, 32);   // combine row halves (m 0-7 with 8-15)
    colsum += s;
    __syncthreads();              // protect As before next chunk overwrites
  }

  // each (expert, column) owned by exactly one wave: plain store
  if (half == 0)
    hsum[(size_t)e * H_DIM + h0 + col] = colsum;
}

// ---------------------------------------------------------------------------
// Kernel 4: down projection of the 8 accumulated H-vectors into output rows
// 0..7 (the expert-id-indexed scatter of the reference).
// ---------------------------------------------------------------------------
__global__ void moe_down(const float* __restrict__ hsum,
                         const float* __restrict__ w_down,
                         float* __restrict__ out) {
  const int e = blockIdx.y;
  const int d = blockIdx.x * blockDim.x + threadIdx.x;   // 0..D-1
  const float* hs = hsum + (size_t)e * H_DIM;
  const float* wd = w_down + (size_t)e * H_DIM * D_DIM + d;
  float acc = 0.0f;
#pragma unroll 4
  for (int h = 0; h < H_DIM; ++h)
    acc += hs[h] * wd[(size_t)h * D_DIM];
  out[(size_t)e * D_DIM + d] = acc;
}

// ---------------------------------------------------------------------------
extern "C" void kernel_launch(void* const* d_in, const int* in_sizes, int n_in,
                              void* d_out, int out_size, void* d_ws, size_t ws_size,
                              hipStream_t stream) {
  const float* x      = (const float*)d_in[0];   // [B,S,D]
  const float* gate_w = (const float*)d_in[1];   // [D,E]
  const float* w_gate = (const float*)d_in[2];   // [E,D,H]
  const float* w_up   = (const float*)d_in[3];   // [E,D,H]
  const float* w_down = (const float*)d_in[4];   // [E,H,D]
  float* out = (float*)d_out;                    // final [B,S,D] then aux [1]

  // workspace layout
  char* ws = (char*)d_ws;
  int*   cnt   = (int*)ws;                                       // [E]
  int*   f_cnt = cnt + E_DIM;                                    // [E]
  float* p_sum = (float*)(f_cnt + E_DIM);                        // [E]
  int*   list  = (int*)(ws + 128);                               // [E][T]
  float* wl    = (float*)(ws + 128 + (size_t)E_DIM * T_TOK * 4); // [E][T]
  float* hsum  = (float*)(ws + 128 + (size_t)2 * E_DIM * T_TOK * 4); // [E][H]

  moe_zero<<<4096, 256, 0, stream>>>(out, OUT_N, cnt, f_cnt, p_sum);

  moe_gate<<<T_TOK / 8, 256, 0, stream>>>(x, gate_w, cnt, f_cnt, p_sum,
                                          list, wl);

  moe_aux<<<1, 32, 0, stream>>>(f_cnt, p_sum, out + (size_t)T_TOK * D_DIM);

  // 288KB dynamic LDS: Bg(128K) + Bu(128K) + As(32K)  (CDNA5: 320KB per WGP)
  size_t smem = ((size_t)2 * D_DIM * TILE_N + (size_t)16 * D_DIM) * 2u;
  dim3 grid_g(H_DIM / TILE_N, E_DIM);
  moe_expert_gemm<<<grid_g, 128, smem, stream>>>(x, w_gate, w_up, cnt,
                                                 list, wl, hsum);

  moe_down<<<dim3(D_DIM / 256, E_DIM), 256, 0, stream>>>(hsum, w_down, out);
}
